// NeuralSDE_32512902431127
// MI455X (gfx1250) — compile-verified, matrix operand order checked by reading, not run on hardware
//
#include <hip/hip_runtime.h>
#include <math.h>

// ---------------- Problem constants (from reference) ----------------
#define BSZ   256   // batch
#define HID   128   // hidden
#define NOI   32    // noise dim
#define WID   256   // MLP width
#define NHID  3     // depth-1 hidden layers
#define MT    16    // batch rows per workgroup (== WMMA M)
#define NWG   (BSZ / MT)   // 16 workgroups

// LDS pitches (elements); row start & +16-elem half-offset stay 32B aligned
#define P128  144
#define P256  272

// bf16 weight arena offsets in d_ws (element offsets)
#define OFF_MUWIN   ((size_t)0)
#define OFF_MUWH    ((size_t)32768)
#define OFF_MUWOUT  ((size_t)229376)
#define OFF_SFWIN   ((size_t)262144)
#define OFF_SFWH    ((size_t)294912)
#define OFF_SFWOUT  ((size_t)491520)

typedef __attribute__((ext_vector_type(16))) __bf16 v16bf;
typedef __attribute__((ext_vector_type(8)))  float  v8f;

__device__ __forceinline__ __bf16 f2bf(float x) {
    unsigned u = __float_as_uint(x);
    u += 0x7fffu + ((u >> 16) & 1u);
    unsigned short s = (unsigned short)(u >> 16);
    __bf16 r;
    __builtin_memcpy(&r, &s, 2);
    return r;
}

__device__ __forceinline__ float fast_rcp(float x) { return __builtin_amdgcn_rcpf(x); }

// tanh(x) = sign(x) * (1-e)/(1+e), e = exp(-2|x|)  -- no div sequence
__device__ __forceinline__ float fast_tanh(float x) {
    float e = __expf(-2.0f * fabsf(x));
    float r = (1.0f - e) * fast_rcp(1.0f + e);
    return copysignf(r, x);
}
// 0.909 * x * sigmoid(x)
__device__ __forceinline__ float lipswish(float x) {
    return 0.909f * x * fast_rcp(1.0f + __expf(-x));
}

__device__ __forceinline__ unsigned rotl32(unsigned x, int r) {
    return (x << r) | (x >> (32 - r));
}

// threefry2x32-20
__device__ __forceinline__ void threefry2x32(unsigned k0, unsigned k1,
                                             unsigned c0, unsigned c1,
                                             unsigned& o0, unsigned& o1) {
    const int R[8] = {13, 15, 26, 6, 17, 29, 16, 24};
    unsigned ks0 = k0, ks1 = k1, ks2 = 0x1BD11BDAu ^ k0 ^ k1;
    unsigned x0 = c0 + ks0, x1 = c1 + ks1;
#pragma unroll
    for (int blk = 0; blk < 5; ++blk) {
#pragma unroll
        for (int r = 0; r < 4; ++r) {
            x0 += x1;
            x1 = rotl32(x1, R[(blk & 1) * 4 + r]);
            x1 ^= x0;
        }
        unsigned a = (blk + 1) % 3 == 0 ? ks0 : ((blk + 1) % 3 == 1 ? ks1 : ks2);
        unsigned b = (blk + 2) % 3 == 0 ? ks0 : ((blk + 2) % 3 == 1 ? ks1 : ks2);
        x0 += a;
        x1 += b + (unsigned)(blk + 1);
    }
    o0 = x0; o1 = x1;
}

__device__ __forceinline__ v8f wmma_bf16(v16bf a, v16bf b, v8f c) {
    return __builtin_amdgcn_wmma_f32_16x16x32_bf16(false, a, false, b, (short)0, c,
                                                   false, false);
}

// One 16x16 output tile, K = 32*KB.  Unroll capped at 4 to keep VGPRs < 256.
template <int KB>
__device__ __forceinline__ v8f gemm_tile(const __bf16* ap, const __bf16* bp) {
    v8f acc = {};
#pragma unroll 4
    for (int kb = 0; kb < KB; ++kb) {
        v16bf a = *(const v16bf*)(ap + kb * 32);
        v16bf b = *(const v16bf*)(bp + kb * 32);
        acc = wmma_bf16(a, b, acc);
    }
    return acc;
}

// ---------------- prep: convert fp32 weights to bf16 arena ----------------
__global__ void prep_bf16(const float* __restrict__ muWin, const float* __restrict__ muWh,
                          const float* __restrict__ muWout, const float* __restrict__ sfWin,
                          const float* __restrict__ sfWh, const float* __restrict__ sfWout,
                          __bf16* __restrict__ wbf) {
    size_t i = (size_t)blockIdx.x * blockDim.x + threadIdx.x;
    size_t stride = (size_t)gridDim.x * blockDim.x;
    for (size_t e = i; e < (size_t)256 * 128; e += stride)
        wbf[OFF_MUWIN + e] = f2bf(muWin[(e >> 7) * 129 + 1 + (e & 127)]);
    for (size_t e = i; e < (size_t)3 * 256 * 256; e += stride)
        wbf[OFF_MUWH + e] = f2bf(muWh[e]);
    for (size_t e = i; e < (size_t)128 * 256; e += stride)
        wbf[OFF_MUWOUT + e] = f2bf(muWout[e]);
    for (size_t e = i; e < (size_t)256 * 128; e += stride)
        wbf[OFF_SFWIN + e] = f2bf(sfWin[(e >> 7) * 129 + 1 + (e & 127)]);
    for (size_t e = i; e < (size_t)3 * 256 * 256; e += stride)
        wbf[OFF_SFWH + e] = f2bf(sfWh[e]);
    for (size_t e = i; e < (size_t)4096 * 256; e += stride)
        wbf[OFF_SFWOUT + e] = f2bf(sfWout[e]);
}

// ---------------- persistent SDE scan kernel ----------------
// grid = 16 WGs (one 16-row batch tile each), block = 256 threads = 8 wave32s.
__global__ __launch_bounds__(256)
void sde_scan(const float* __restrict__ y0,
              const float* __restrict__ t0p, const float* __restrict__ dtp,
              const float* __restrict__ muWin_f, const float* __restrict__ mu_b_in,
              const float* __restrict__ mu_b_h, const float* __restrict__ mu_b_out,
              const float* __restrict__ sfWin_f, const float* __restrict__ sf_b_in,
              const float* __restrict__ sf_b_h, const float* __restrict__ sf_b_out,
              const int* __restrict__ Tp, const int* __restrict__ seedp,
              const __bf16* __restrict__ wbf, float* __restrict__ out) {
    __shared__ __bf16 s_ybf[MT * P128];
    __shared__ __bf16 s_hmu[2][MT * P256];
    __shared__ __bf16 s_gsf[2][MT * P256];
    __shared__ float  s_y[MT * HID];
    __shared__ float  s_yacc[MT * HID];
    __shared__ float  s_bm[MT * NOI];

    const int tid   = threadIdx.x;
    const int lane  = tid & 31;
    const int wg    = blockIdx.x;
    const int l15   = lane & 15;
    const int khalf = (lane >> 4) << 4;   // +16 K elements for upper half-wave
    const int mhalf = (lane >> 4) << 3;   // +8 M rows for C/D layout
    // scalar (SGPR) wave id -> wave-uniform branches compile to s_cbranch
    const int wv = __builtin_amdgcn_readfirstlane(tid >> 5);

    const float t0 = t0p[0];
    const float dt = dtp[0];
    const int   T  = Tp[0];
    const unsigned seed = (unsigned)seedp[0];
    const float sqrtdt = sqrtf(dt);

    const __bf16* muWin  = wbf + OFF_MUWIN;
    const __bf16* muWh   = wbf + OFF_MUWH;
    const __bf16* muWout = wbf + OFF_MUWOUT;
    const __bf16* sfWin  = wbf + OFF_SFWIN;
    const __bf16* sfWh   = wbf + OFF_SFWH;
    const __bf16* sfWout = wbf + OFF_SFWOUT;

    // load this WG's 16-row y0 slice (8 consecutive floats / thread)
    {
        const int base = tid * 8;
#pragma unroll
        for (int i = 0; i < 8; ++i)
            s_y[base + i] = y0[(size_t)wg * MT * HID + base + i];
    }
    __syncthreads();

    for (int step = 0; step < T; ++step) {
        const float t = t0 + (float)step * dt;

        // ---- phase 0: Brownian increments + bf16(y) + yacc init ----
        {
            int m = tid >> 4, pr = tid & 15;
            unsigned o0, o1;
            threefry2x32(seed, (unsigned)step,
                         (unsigned)((wg * MT + m) * 16 + pr), 0x03579BDFu, o0, o1);
            float u1 = ((float)(o0 >> 8) + 0.5f) * (1.0f / 16777216.0f);
            float u2 = ((float)(o1 >> 8) + 0.5f) * (1.0f / 16777216.0f);
            float rr = sqrtf(fmaxf(-2.0f * __logf(u1), 0.0f)) * sqrtdt;
            float sn, cs;
            __sincosf(6.28318530718f * u2, &sn, &cs);
            s_bm[m * NOI + 2 * pr]     = rr * cs;
            s_bm[m * NOI + 2 * pr + 1] = rr * sn;
        }
        {
            const int base = tid * 8;          // 8 consecutive elems -> b128 traffic
            const int m = base >> 7, h0 = base & 127;
#pragma unroll
            for (int i = 0; i < 8; ++i) {
                float v = s_y[base + i];
                s_ybf[m * P128 + h0 + i] = f2bf(v);
                s_yacc[base + i] = v;
            }
        }
        __syncthreads();

        // ---- layer 1: K=128. waves 0-3 -> mu (ReLU), waves 4-7 -> sf (LipSwish) ----
        if (wv < 4) {
            for (int tI = wv * 4; tI < wv * 4 + 4; ++tI) {
                const int ncol = tI * 16 + l15;
                v8f acc = gemm_tile<4>(s_ybf + l15 * P128 + khalf,
                                       muWin + (size_t)ncol * 128 + khalf);
                const float bias = mu_b_in[ncol] + t * muWin_f[(size_t)ncol * 129];
#pragma unroll
                for (int r = 0; r < 8; ++r)
                    s_hmu[0][(r + mhalf) * P256 + ncol] = f2bf(fmaxf(acc[r] + bias, 0.0f));
            }
        } else {
            for (int tI = (wv - 4) * 4; tI < (wv - 4) * 4 + 4; ++tI) {
                const int ncol = tI * 16 + l15;
                v8f acc = gemm_tile<4>(s_ybf + l15 * P128 + khalf,
                                       sfWin + (size_t)ncol * 128 + khalf);
                const float bias = sf_b_in[ncol] + t * sfWin_f[(size_t)ncol * 129];
#pragma unroll
                for (int r = 0; r < 8; ++r)
                    s_gsf[0][(r + mhalf) * P256 + ncol] = f2bf(lipswish(acc[r] + bias));
            }
        }
        __syncthreads();

        // ---- hidden layers: K=256, ping-pong ----
        for (int j = 0; j < NHID; ++j) {
            const int src = j & 1, dsti = 1 - src;
            if (wv < 4) {
                const __bf16* W = muWh + (size_t)j * WID * WID;
                for (int tI = wv * 4; tI < wv * 4 + 4; ++tI) {
                    const int ncol = tI * 16 + l15;
                    v8f acc = gemm_tile<8>(s_hmu[src] + l15 * P256 + khalf,
                                           W + (size_t)ncol * 256 + khalf);
                    const float bias = mu_b_h[j * WID + ncol];
#pragma unroll
                    for (int r = 0; r < 8; ++r)
                        s_hmu[dsti][(r + mhalf) * P256 + ncol] =
                            f2bf(fmaxf(acc[r] + bias, 0.0f));
                }
            } else {
                const __bf16* W = sfWh + (size_t)j * WID * WID;
                for (int tI = (wv - 4) * 4; tI < (wv - 4) * 4 + 4; ++tI) {
                    const int ncol = tI * 16 + l15;
                    v8f acc = gemm_tile<8>(s_gsf[src] + l15 * P256 + khalf,
                                           W + (size_t)ncol * 256 + khalf);
                    const float bias = sf_b_h[j * WID + ncol];
#pragma unroll
                    for (int r = 0; r < 8; ++r)
                        s_gsf[dsti][(r + mhalf) * P256 + ncol] =
                            f2bf(lipswish(acc[r] + bias));
                }
            }
            __syncthreads();
        }
        const __bf16* hF = s_hmu[1];   // final buffers after 3 ping-pongs
        const __bf16* gF = s_gsf[1];

        // ---- drift output: N=128, tile wv per wave; yacc += tanh(.)*dt ----
        {
            const int ncol = wv * 16 + l15;
            v8f acc = gemm_tile<8>(hF + l15 * P256 + khalf,
                                   muWout + (size_t)ncol * 256 + khalf);
            const float bias = mu_b_out[ncol];
#pragma unroll
            for (int r = 0; r < 8; ++r)
                atomicAdd(&s_yacc[(r + mhalf) * HID + ncol],
                          fast_tanh(acc[r] + bias) * dt);
        }

        // ---- diffusion output fused with einsum: N=4096, 16 h-pairs per wave ----
        // Brownian factors per lane are step-invariant across tiles: hoist them.
        float bmlo[8], bmhi[8];
#pragma unroll
        for (int r = 0; r < 8; ++r) {
            bmlo[r] = s_bm[(r + mhalf) * NOI + l15];        // even tile: nidx = l15
            bmhi[r] = s_bm[(r + mhalf) * NOI + 16 + l15];   // odd tile:  nidx = 16+l15
        }
        const __bf16* apg = gF + l15 * P256 + khalf;
        for (int p = 0; p < 16; ++p) {
            const int tI   = wv * 32 + p * 2;    // even/odd tile pair -> same h column
            const int n0   = tI * 16;
            const int ncol = n0 + l15;
            if (p + 1 < 16)
                __builtin_prefetch(sfWout + (size_t)(ncol + 32) * 256 + khalf, 0, 1);
            v8f acc0 = gemm_tile<8>(apg, sfWout + (size_t)ncol * 256 + khalf);
            v8f acc1 = gemm_tile<8>(apg, sfWout + (size_t)(ncol + 16) * 256 + khalf);
            const float b0 = sf_b_out[ncol];
            const float b1 = sf_b_out[ncol + 16];
            const int h = n0 >> 5;
#pragma unroll
            for (int r = 0; r < 8; ++r) {
                const int m = r + mhalf;
                float v = fast_tanh(acc0[r] + b0) * bmlo[r]
                        + fast_tanh(acc1[r] + b1) * bmhi[r];
                v += __shfl_xor(v, 1);
                v += __shfl_xor(v, 2);
                v += __shfl_xor(v, 4);
                v += __shfl_xor(v, 8);
                if ((lane & 15) == 0) atomicAdd(&s_yacc[m * HID + h], v);
            }
        }
        __syncthreads();

        // ---- commit: y = yacc; write ys[step] slice (8 consecutive / thread) ----
        {
            const int base = tid * 8;
            float* op = out + (size_t)step * BSZ * HID + (size_t)wg * MT * HID + base;
#pragma unroll
            for (int i = 0; i < 8; ++i) {
                float v = s_yacc[base + i];
                s_y[base + i] = v;
                op[i] = v;
            }
        }
        __syncthreads();
    }
}

extern "C" void kernel_launch(void* const* d_in, const int* in_sizes, int n_in,
                              void* d_out, int out_size, void* d_ws, size_t ws_size,
                              hipStream_t stream) {
    (void)in_sizes; (void)n_in; (void)out_size; (void)ws_size;
    const float* y0       = (const float*)d_in[0];
    const float* t0       = (const float*)d_in[1];
    const float* dt       = (const float*)d_in[2];
    const float* muWin    = (const float*)d_in[3];
    const float* mu_b_in  = (const float*)d_in[4];
    const float* muWh     = (const float*)d_in[5];
    const float* mu_b_h   = (const float*)d_in[6];
    const float* muWout   = (const float*)d_in[7];
    const float* mu_b_out = (const float*)d_in[8];
    const float* sfWin    = (const float*)d_in[9];
    const float* sf_b_in  = (const float*)d_in[10];
    const float* sfWh     = (const float*)d_in[11];
    const float* sf_b_h   = (const float*)d_in[12];
    const float* sfWout   = (const float*)d_in[13];
    const float* sf_b_out = (const float*)d_in[14];
    const int*   T        = (const int*)d_in[15];
    const int*   seed     = (const int*)d_in[17];

    __bf16* wbf = (__bf16*)d_ws;

    prep_bf16<<<512, 256, 0, stream>>>(muWin, muWh, muWout, sfWin, sfWh, sfWout, wbf);
    sde_scan<<<NWG, 256, 0, stream>>>(y0, t0, dt,
                                      muWin, mu_b_in, mu_b_h, mu_b_out,
                                      sfWin, sf_b_in, sf_b_h, sf_b_out,
                                      T, seed, wbf, (float*)d_out);
}
